// RNN_67370857005355
// MI455X (gfx1250) — matrix operand order repaired and checked by reading, not compiled, not run on hardware
//
#include <hip/hip_runtime.h>
#include <hip/hip_bf16.h>
#include <math.h>

// Problem constants (match reference)
#define B_DIM 256
#define T_DIM 512
#define I_DIM 2
#define H_DIM 512
#define O_DIM 512
#define ALPHA_F 0.1f

typedef __attribute__((ext_vector_type(16))) __bf16 bf16x16;
typedef __attribute__((ext_vector_type(8)))  __bf16 bf16x8;
typedef __attribute__((ext_vector_type(8)))  float  f32x8;

__device__ __forceinline__ unsigned short f32_to_bf16_rne(float f) {
  unsigned int u = __float_as_uint(f);
  u += 0x7fffu + ((u >> 16) & 1u);   // round-to-nearest-even
  return (unsigned short)(u >> 16);
}

__device__ __forceinline__ bf16x16 cat8(bf16x8 lo, bf16x8 hi) {
  return __builtin_shufflevector(lo, hi, 0,1,2,3,4,5,6,7,8,9,10,11,12,13,14,15);
}

// Branch-free tanh via v_exp_f32 + v_rcp_f32: tanh(x) = 1 - 2/(e^{2x}+1).
__device__ __forceinline__ float fast_tanh(float x) {
  float xc = fminf(fmaxf(x, -9.0f), 9.0f);
  float e  = __builtin_amdgcn_exp2f(xc * 2.8853900817779268f);  // 2*log2(e)
  return 1.0f - 2.0f * __builtin_amdgcn_rcpf(e + 1.0f);
}

// ---------------------------------------------------------------------------
// Kernel 0: one-time f32 -> bf16 conversion of W_rec and W_out into d_ws.
// ---------------------------------------------------------------------------
__global__ void convert_weights(const float* __restrict__ Wrec,
                                const float* __restrict__ Wout,
                                unsigned short* __restrict__ wrec_bf,
                                unsigned short* __restrict__ wout_bf) {
  const int n = H_DIM * H_DIM;
  for (int i = blockIdx.x * blockDim.x + threadIdx.x; i < n;
       i += gridDim.x * blockDim.x) {
    wrec_bf[i] = f32_to_bf16_rne(Wrec[i]);
    wout_bf[i] = f32_to_bf16_rne(Wout[i]);
  }
}

// ---------------------------------------------------------------------------
// Kernel 1: persistent recurrent scan. One block per 16 batch rows -> the
// whole T=512 recurrence is block-private (no global sync). h kept in LDS as
// f32 master + bf16 WMMA copy. 32 waves; each wave owns a 16-column N-slice
// and runs a K=512 dual-chain of v_wmma_f32_16x16x32_bf16 per step.
// W_rec fragments are streamed from L2 with a depth-1 rotating prefetch that
// is deliberately kept non-hoistable (#pragma unroll 1, prologue inside the
// t-loop) so the compiler cannot LICM the whole B slice into spilled VGPRs.
// ---------------------------------------------------------------------------
__global__ __launch_bounds__(1024) void rnn_scan(
    const float* __restrict__ inputs,            // (B, T, 2)
    const float* __restrict__ init_ctx,          // (B, 512)
    const float* __restrict__ W_in,              // (512, 2)
    const float* __restrict__ b_in,              // (512)
    const unsigned short* __restrict__ Wrec_bf,  // (512, 512) bf16 row-major
    const float* __restrict__ b_rec,             // (512)
    const float* __restrict__ W_h_init,          // (512, 512)
    float* __restrict__ hidden)                  // (B, T, 512) f32 out
{
  __shared__ alignas(32) float          h_f32[16][H_DIM];     // 32 KB
  __shared__ alignas(32) unsigned short h_bf [16][H_DIM];     // 16 KB
  __shared__ float w_in_s[H_DIM][2];                          // 4 KB
  __shared__ float b_sum_s[H_DIM];                            // b_in + b_rec
  extern __shared__ float x_all[];                            // 64 KB dynamic:
                                                              // [16][T][2] f32

  const int tid  = threadIdx.x;
  const int lane = tid & 31;
  const int wave = tid >> 5;          // 0..31
  const int m0   = blockIdx.x * 16;   // batch-row base

  // ---- async-stage the block's whole x slice: global -> LDS ---------------
  // Region is contiguous: inputs + m0*T*2, 16*T*2 floats = 65536 bytes.
  {
    const unsigned long long src =
        (unsigned long long)(inputs + (size_t)m0 * T_DIM * I_DIM);
    const unsigned lds0 = (unsigned)(unsigned long long)(void*)x_all;
    #pragma unroll
    for (int c = 0; c < 4; ++c) {
      const unsigned off = (unsigned)(tid * 16 + c * 16384);  // bytes
      unsigned lds_addr = lds0 + off;
      unsigned long long gaddr = src + off;
      asm volatile("global_load_async_to_lds_b128 %0, %1, off"
                   :: "v"(lds_addr), "v"(gaddr) : "memory");
    }
  }

  // ---- stage small parameters (overlaps with async copy) ------------------
  if (tid < H_DIM) {
    w_in_s[tid][0] = W_in[tid * 2 + 0];
    w_in_s[tid][1] = W_in[tid * 2 + 1];
    b_sum_s[tid]   = b_in[tid] + b_rec[tid];
  }

  // ---- h0 = init_ctx @ W_h_init^T for this block's 16 rows ----------------
  {
    const int col   = tid & (H_DIM - 1);          // 0..511
    const int mhalf = tid >> 9;                   // 0/1 -> rows 0-7 / 8-15
    const float* wrow = W_h_init + (size_t)col * O_DIM;
    for (int m = 8 * mhalf; m < 8 * mhalf + 8; ++m) {
      const float* irow = init_ctx + (size_t)(m0 + m) * O_DIM;
      float acc = 0.0f;
      #pragma unroll 4
      for (int o = 0; o < O_DIM; ++o) acc = fmaf(irow[o], wrow[o], acc);
      h_f32[m][col] = acc;
      h_bf[m][col]  = f32_to_bf16_rne(acc);
    }
  }
  asm volatile("s_wait_asynccnt 0x0" ::: "memory");  // x slice landed in LDS
  __syncthreads();

  const int nBase = wave * 16;        // this wave's 16 output columns
  const int khalf = lane >> 4;        // 0 or 1 (lane half)
  const int lmod  = lane & 15;
  // This lane's W_rec row base (B[k][n] = W_rec[n][k]; 32B contiguous/lane).
  const unsigned short* bptr =
      Wrec_bf + (size_t)(nBase + lmod) * H_DIM + 16 * khalf;
  const float wi0 = w_in_s[nBase + lmod][0];
  const float wi1 = w_in_s[nBase + lmod][1];
  const float bs  = b_sum_s[nBase + lmod];

  for (int t = 0; t < T_DIM; ++t) {
    __syncthreads();  // h_bf from previous step visible to all waves

    // ---- C += A(16x512 bf16 from LDS) * B(512x16, W_rec^T rows) -----------
    // Dual accumulator chains; rotating depth-1 prefetch of B fragments.
    f32x8 accE = {0.f, 0.f, 0.f, 0.f, 0.f, 0.f, 0.f, 0.f};
    f32x8 accO = {0.f, 0.f, 0.f, 0.f, 0.f, 0.f, 0.f, 0.f};
    bf16x16 b0 = *reinterpret_cast<const bf16x16*>(bptr);
    bf16x16 b1 = *reinterpret_cast<const bf16x16*>(bptr + 32);
    #pragma unroll 1
    for (int kc = 0; kc < H_DIM; kc += 64) {
      const int kn = (kc + 64) & (H_DIM - 1);     // wraps on last iter (unused)
      bf16x16 p0 = *reinterpret_cast<const bf16x16*>(bptr + kn);
      bf16x16 p1 = *reinterpret_cast<const bf16x16*>(bptr + kn + 32);
      // A as loaded here (k order matched by B): lanes 0-15 hold K=+0..15,
      // lanes 16-31 hold K=+16..31; row M = lane%16.
      bf16x16 a0 = cat8(
          *reinterpret_cast<const bf16x8*>(&h_bf[lmod][kc + 16 * khalf]),
          *reinterpret_cast<const bf16x8*>(&h_bf[lmod][kc + 8 + 16 * khalf]));
      bf16x16 a1 = cat8(
          *reinterpret_cast<const bf16x8*>(&h_bf[lmod][kc + 32 + 16 * khalf]),
          *reinterpret_cast<const bf16x8*>(&h_bf[lmod][kc + 40 + 16 * khalf]));
      accE = __builtin_amdgcn_wmma_f32_16x16x32_bf16(
          false, a0, false, b0, (short)0, accE, false, false);
      accO = __builtin_amdgcn_wmma_f32_16x16x32_bf16(
          false, a1, false, b1, (short)0, accO, false, false);
      b0 = p0; b1 = p1;
    }
    const f32x8 acc = accE + accO;
    __syncthreads();  // all waves done reading h_bf for step t

    // ---- leaky tanh update; each lane owns 8 C elements --------------------
    // C layout: VGPR r -> M = r + 8*khalf, N = nBase + lane%16
    const int n = nBase + lmod;
    #pragma unroll
    for (int r = 0; r < 8; ++r) {
      const int m = r + 8 * khalf;
      const float x0 = x_all[(m * T_DIM + t) * I_DIM + 0];
      const float x1 = x_all[(m * T_DIM + t) * I_DIM + 1];
      const float pre = acc[r] + x0 * wi0 + x1 * wi1 + bs;
      const float hn =
          (1.0f - ALPHA_F) * h_f32[m][n] + ALPHA_F * fast_tanh(pre);
      h_f32[m][n] = hn;
      h_bf[m][n]  = f32_to_bf16_rne(hn);
      hidden[((size_t)(m0 + m) * T_DIM + t) * H_DIM + n] = hn;
    }
    // next iteration's top barrier makes these h writes visible
  }
}

// ---------------------------------------------------------------------------
// Kernel 2: outputs = hidden @ W_out^T.  M = B*T = 131072, N = K = 512.
// Block: 256 threads / 8 waves, 32-row A tile staged to LDS as bf16;
// each wave computes a 16x128 C tile (8 independent accumulator chains).
// ---------------------------------------------------------------------------
__global__ __launch_bounds__(256) void out_gemm(
    const float* __restrict__ hidden,            // (B*T, 512) f32
    const unsigned short* __restrict__ Wout_bf,  // (512, 512) bf16
    float* __restrict__ out)                     // (B*T, 512) f32
{
  __shared__ alignas(32) unsigned short a_s[32][H_DIM];  // 32 KB

  const int tid  = threadIdx.x;
  const int lane = tid & 31;
  const int wave = tid >> 5;                 // 0..7
  const size_t m0 = (size_t)blockIdx.x * 32;

  // stage + convert 32 x 512 A tile
  for (int i = tid; i < 32 * H_DIM; i += 256) {
    const int r = i >> 9, c = i & (H_DIM - 1);
    a_s[r][c] = f32_to_bf16_rne(hidden[(m0 + r) * H_DIM + c]);
  }
  __syncthreads();

  const int mSub  = wave & 1;                // which 16-row half
  const int nBase = (wave >> 1) * 128;       // 128-column slice
  const int khalf = lane >> 4;
  const int lmod  = lane & 15;

  f32x8 acc[8];
  #pragma unroll
  for (int j = 0; j < 8; ++j)
    acc[j] = (f32x8){0.f, 0.f, 0.f, 0.f, 0.f, 0.f, 0.f, 0.f};

  #pragma unroll 2
  for (int kc = 0; kc < H_DIM; kc += 32) {
    bf16x16 a = cat8(
        *reinterpret_cast<const bf16x8*>(&a_s[mSub * 16 + lmod][kc + 16 * khalf]),
        *reinterpret_cast<const bf16x8*>(
            &a_s[mSub * 16 + lmod][kc + 8 + 16 * khalf]));
    #pragma unroll
    for (int j = 0; j < 8; ++j) {
      bf16x16 b = *reinterpret_cast<const bf16x16*>(
          Wout_bf + (size_t)(nBase + j * 16 + lmod) * H_DIM + kc + 16 * khalf);
      acc[j] = __builtin_amdgcn_wmma_f32_16x16x32_bf16(
          false, a, false, b, (short)0, acc[j], false, false);
    }
  }

  #pragma unroll
  for (int j = 0; j < 8; ++j) {
    const int n = nBase + j * 16 + lmod;
    #pragma unroll
    for (int r = 0; r < 8; ++r) {
      const int m = mSub * 16 + r + 8 * khalf;
      out[(m0 + m) * H_DIM + n] = acc[j][r];
    }
  }
}

// ---------------------------------------------------------------------------
extern "C" void kernel_launch(void* const* d_in, const int* in_sizes, int n_in,
                              void* d_out, int out_size, void* d_ws, size_t ws_size,
                              hipStream_t stream) {
  (void)in_sizes; (void)n_in; (void)out_size; (void)ws_size;

  const float* inputs       = (const float*)d_in[0];  // (B,T,2)
  const float* init_context = (const float*)d_in[1];  // (B,512)
  const float* W_in         = (const float*)d_in[2];  // (512,2)
  const float* b_in         = (const float*)d_in[3];  // (512)
  const float* W_rec        = (const float*)d_in[4];  // (512,512)
  const float* b_rec        = (const float*)d_in[5];  // (512)
  const float* W_out        = (const float*)d_in[6];  // (512,512)
  const float* W_h_init     = (const float*)d_in[7];  // (512,512)

  float* hidden  = (float*)d_out;                               // (B,T,H)
  float* outputs = hidden + (size_t)B_DIM * T_DIM * H_DIM;      // (B,T,O)

  unsigned short* wrec_bf = (unsigned short*)d_ws;              // 512 KB
  unsigned short* wout_bf = wrec_bf + (size_t)H_DIM * H_DIM;    // 512 KB

  convert_weights<<<512, 256, 0, stream>>>(W_rec, W_out, wrec_bf, wout_bf);

  // 64 KB dynamic LDS for the per-block input slice (async-staged).
  rnn_scan<<<B_DIM / 16, 1024, 16 * T_DIM * I_DIM * sizeof(float), stream>>>(
      inputs, init_context, W_in, b_in, wrec_bf, b_rec, W_h_init, hidden);

  out_gemm<<<(B_DIM * T_DIM) / 32, 256, 0, stream>>>(hidden, wout_bf, outputs);
}
